// Decoder_5068061409853
// MI455X (gfx1250) — matrix operand order repaired and checked by reading, not compiled, not run on hardware
//
#include <hip/hip_runtime.h>
#include <math.h>

// ---------------- constants ----------------
#define NBATCH 4
#define LATD   512
#define NSTEP  6
#define NF     32
#define CIN    384
#define CGRP   96
#define NGRP   4
#define HWPIX  4096
#define WSZC   36864          // CIN*CG
#define SEGC   37248          // WSZ + CIN
#define KTOTC  111744         // 3*SEG
#define DHID   1024
#define NTILE_KS 6984         // KTOT/16
#define SLAB   524288         // B*NF*HW

typedef __attribute__((ext_vector_type(16))) __bf16 v16bf;
typedef __attribute__((ext_vector_type(8)))  float  v8f;

// Native conversion: lets the backend emit packed f32->bf16 converts instead
// of the integer-ALU round-to-nearest-even bit trick.
__device__ __forceinline__ __bf16 f2bf(float x) { return (__bf16)x; }

__device__ __forceinline__ v8f wmma_bf(v16bf a, v16bf b, v8f c) {
  return __builtin_amdgcn_wmma_f32_16x16x32_bf16(false, a, false, b, (short)0, c,
                                                 false, false);
}

// A fragment: 16x32 tile of row-major f32 matrix, converted to bf16.
// Layout (ISA 7.12.2): lanes 0-15 rows M=0..15; vgpr i<4 -> K=2i,2i+1 (+8 for
// lane half 1); vgpr i>=4 -> K=16+2(i-4),... (+8 for half 1).
__device__ __forceinline__ v16bf load_a(const float* A, int lda, int lr, int hl) {
  v16bf a;
  const float* rp = A + lr * lda;
#pragma unroll
  for (int i = 0; i < 8; ++i) {
    int kk = (i < 4 ? 2 * i : 16 + 2 * (i - 4)) + hl * 8;
    a[2 * i]     = f2bf(rp[kk]);
    a[2 * i + 1] = f2bf(rp[kk + 1]);
  }
  return a;
}

// B fragment: 32x16 tile of row-major (K x N) f32 matrix.
// Same K order as load_a: vgpr v<4 -> K=2v,2v+1; v>=4 -> K=16+2(v-4),... (+8 for lane half 1).
__device__ __forceinline__ v16bf load_b(const float* B, int ldb, int lr, int hl) {
  v16bf b;
#pragma unroll
  for (int v = 0; v < 8; ++v) {
    int kk = (v < 4 ? 2 * v : 16 + 2 * (v - 4)) + hl * 8;
    b[2 * v]     = f2bf(B[kk * ldb + lr]);
    b[2 * v + 1] = f2bf(B[(kk + 1) * ldb + lr]);
  }
  return b;
}

#define ZERO8 {0.f,0.f,0.f,0.f,0.f,0.f,0.f,0.f}

// ---------------- sobel weights (constants in reference) ----------------
__global__ void k_sobelw(float* wbuf) {
  if (threadIdx.x != 0) return;
  const int ksz[4] = {3, 5, 9, 17};
  int s = blockIdx.x;
  int k = ksz[s];
  float* w = wbuf + s * 289;
  float sn[17], cs[17];
  for (int i = 0; i < k; ++i) {
    float t = -1.f + 2.f * i / (float)(k - 1);
    sn[i] = sinf(t * 1.57079632679489662f);
    cs[i] = cosf(t * 1.57079632679489662f);
  }
  float ssum = 0.f;
  for (int i = 0; i < k; ++i)
    for (int j = 0; j < k; ++j) {
      float v = cs[i] * sn[j];
      w[i * k + j] = v;
      ssum += fabsf(v);
    }
  float inv = 1.f / ssum;
  for (int i = 0; i < k * k; ++i) w[i] *= inv;
}

// ---------------- latent recurrence: all 6 steps up front ----------------
__global__ void k_latseq(const float* __restrict__ lat, const float* __restrict__ W0,
                         const float* __restrict__ b0, const float* __restrict__ W1,
                         const float* __restrict__ b1, float* __restrict__ Lseq) {
  __shared__ float l[NBATCH][LATD];
  __shared__ float tmp[NBATCH][LATD];
  int j = threadIdx.x;
  for (int b = 0; b < NBATCH; ++b) l[b][j] = lat[b * LATD + j];
  __syncthreads();
  for (int t = 0; t < NSTEP; ++t) {
    const float* w0 = W0 + (size_t)t * LATD * LATD;
    const float* w1 = W1 + (size_t)t * LATD * LATD;
    float acc[NBATCH] = {0.f, 0.f, 0.f, 0.f};
    for (int i = 0; i < LATD; ++i) {
      float w = w0[i * LATD + j];
      acc[0] += l[0][i] * w; acc[1] += l[1][i] * w;
      acc[2] += l[2][i] * w; acc[3] += l[3][i] * w;
    }
    float bb = b0[t * LATD + j];
    for (int b = 0; b < NBATCH; ++b) tmp[b][j] = fmaxf(acc[b] + bb, 0.f);
    __syncthreads();
    float d[NBATCH] = {0.f, 0.f, 0.f, 0.f};
    for (int i = 0; i < LATD; ++i) {
      float w = w1[i * LATD + j];
      d[0] += tmp[0][i] * w; d[1] += tmp[1][i] * w;
      d[2] += tmp[2][i] * w; d[3] += tmp[3][i] * w;
    }
    float bv = b1[t * LATD + j];
    for (int b = 0; b < NBATCH; ++b) {
      float ln = l[b][j] + 0.1f * (l[b][j] + d[b] + bv);
      l[b][j] = ln;
      Lseq[(size_t)(t * NBATCH + b) * LATD + j] = ln;
    }
    __syncthreads();
  }
}

// ---------------- H = relu(Lseq @ dyn_W1 + b1), rows 24..31 zero ----------
__global__ void k_hidden(const float* __restrict__ Lseq, const float* __restrict__ W1,
                         const float* __restrict__ b1, float* __restrict__ H) {
  int idx = blockIdx.x * 256 + threadIdx.x;   // 32*1024
  int m = idx >> 10, j = idx & 1023;
  float acc = 0.f;
  if (m < NSTEP * NBATCH) {
    const float* lr = Lseq + (size_t)m * LATD;
    for (int i = 0; i < LATD; ++i) acc += lr[i] * W1[i * DHID + j];
    acc = fmaxf(acc + b1[j], 0.f);
  }
  H[idx] = acc;
}

// ---------------- KS = H @ dyn_W2 + b2 (one pass over 457MB weight) ------
__global__ void k_ksgemm(const float* __restrict__ H, const float* __restrict__ W2,
                         const float* __restrict__ b2, float* __restrict__ KS) {
  int wid  = (blockIdx.x * 256 + threadIdx.x) >> 5;   // 13968 waves
  int lane = threadIdx.x & 31, lr = lane & 15, hl = lane >> 4;
  int nt = wid % NTILE_KS, mt = wid / NTILE_KS;
  int row0 = mt * 16, col0 = nt * 16;
  v8f c = ZERO8;
  for (int k0 = 0; k0 < DHID; k0 += 32) {
    // prefetch next K-slice of the streaming 457MB weight panel
    if (k0 + 32 < DHID)
      __builtin_prefetch(W2 + (size_t)(k0 + 32) * KTOTC + col0 + lr, 0, 1);
    v16bf a = load_a(H + row0 * DHID + k0, DHID, lr, hl);
    v16bf b = load_b(W2 + (size_t)k0 * KTOTC + col0, KTOTC, lr, hl);
    c = wmma_bf(a, b, c);
  }
  int col = col0 + lr;
  float bias = b2[col];
  // validity of the 8 output rows is uniform per lane-half: single branch
  if (row0 + hl * 8 < NSTEP * NBATCH) {
#pragma unroll
    for (int r = 0; r < 8; ++r) {
      int rr = row0 + hl * 8 + r;
      KS[(size_t)rr * KTOTC + col] = c[r] + bias;
    }
  }
}

// ---------------- out_embs[0] = broadcast seed ---------------------------
__global__ void k_seed(const float* __restrict__ seed, float* __restrict__ e0) {
  int idx = blockIdx.x * 256 + threadIdx.x;   // 524288
  e0[idx] = seed[idx & 131071];
}

// ---------------- fused sin_sobel + instance norm ------------------------
__global__ void k_sobelnorm(const float* __restrict__ x, const float* __restrict__ wbuf,
                            float* __restrict__ S) {
  const int ksz[4] = {3, 5, 9, 17};
  const int pad[4] = {1, 2, 4, 8};
  int bc = blockIdx.x;                // b*384 + cout
  int b = bc / CIN, cout = bc % CIN;
  int scale = cout / 96, rem = cout % 96, part = rem >> 5, f = rem & 31;
  int k = ksz[scale], p = pad[scale];
  const float* xin = x + (size_t)(b * NF + f) * HWPIX;
  const float* wk = wbuf + scale * 289;
  int tid = threadIdx.x;
  float vals[16];
  float sum = 0.f, sq = 0.f;
  for (int it = 0; it < 16; ++it) {
    int pix = it * 256 + tid;
    int h = pix >> 6, wv = pix & 63;
    float v;
    if (part == 2) {
      v = xin[pix];
    } else {
      v = 0.f;
      for (int i = 0; i < k; ++i) {
        int hh = h + i - p;
        if (hh < 0 || hh >= 64) continue;
        for (int j = 0; j < k; ++j) {
          int ww = wv + j - p;
          if (ww < 0 || ww >= 64) continue;
          float wt = (part == 0) ? wk[i * k + j] : wk[j * k + i];
          v += wt * xin[hh * 64 + ww];
        }
      }
    }
    vals[it] = v; sum += v; sq += v * v;
  }
  __shared__ float s1[256], s2[256];
  s1[tid] = sum; s2[tid] = sq;
  __syncthreads();
  for (int o = 128; o > 0; o >>= 1) {
    if (tid < o) { s1[tid] += s1[tid + o]; s2[tid] += s2[tid + o]; }
    __syncthreads();
  }
  float m = s1[0] * (1.f / 4096.f);
  float var = s2[0] * (1.f / 4096.f) - m * m;
  float rstd = rsqrtf(var + 1e-5f);
  float* so = S + (size_t)bc * HWPIX;
  for (int it = 0; it < 16; ++it) so[it * 256 + tid] = (vals[it] - m) * rstd;
}

// ---------------- gconv pass 1: g1 = relu(wi@x+bi), xs = ws@x+bs ---------
__global__ void k_gconv1(const float* __restrict__ S, const float* __restrict__ KS,
                         float* __restrict__ G1, float* __restrict__ XS, int t) {
  int wid  = (blockIdx.x * 256 + threadIdx.x) >> 5;   // 16*6*256 waves
  int lane = threadIdx.x & 31, lr = lane & 15, hl = lane >> 4;
  int nt = wid & 255, rest = wid >> 8, mt = rest % 6, bg = rest / 6;
  int b = bg >> 2, g = bg & 3;
  const float* ksrow = KS + (size_t)(t * NBATCH + b) * KTOTC;
  const float* wi  = ksrow + g * CGRP * CGRP;
  const float* wsd = ksrow + 2 * SEGC + g * CGRP * CGRP;
  const float* xg  = S + (size_t)((b * NGRP + g) * CGRP) * HWPIX;
  int row0 = mt * 16, col0 = nt * 16;
  v8f ci = ZERO8, cs = ZERO8;
  for (int k0 = 0; k0 < CGRP; k0 += 32) {
    v16bf bf = load_b(xg + k0 * HWPIX + col0, HWPIX, lr, hl);
    v16bf ai = load_a(wi  + row0 * CGRP + k0, CGRP, lr, hl);
    v16bf as = load_a(wsd + row0 * CGRP + k0, CGRP, lr, hl);
    ci = wmma_bf(ai, bf, ci);
    cs = wmma_bf(as, bf, cs);
  }
  const float* biP = ksrow + WSZC;
  const float* bsP = ksrow + 2 * SEGC + WSZC;
  int col = col0 + lr;
#pragma unroll
  for (int r = 0; r < 8; ++r) {
    int o = row0 + hl * 8 + r;
    size_t oi = ((size_t)(b * NGRP + g) * CGRP + o) * HWPIX + col;
    G1[oi] = fmaxf(ci[r] + biP[g * CGRP + o], 0.f);
    XS[oi] = cs[r] + bsP[g * CGRP + o];
  }
}

// ------- gconv pass 2 + sigmoid gate, sg = s*sigmoid(xs+wo@g1+bo) (in place)
__global__ void k_gconv2(float* __restrict__ S, const float* __restrict__ G1,
                         const float* __restrict__ XS, const float* __restrict__ KS,
                         int t) {
  int wid  = (blockIdx.x * 256 + threadIdx.x) >> 5;
  int lane = threadIdx.x & 31, lr = lane & 15, hl = lane >> 4;
  int nt = wid & 255, rest = wid >> 8, mt = rest % 6, bg = rest / 6;
  int b = bg >> 2, g = bg & 3;
  const float* ksrow = KS + (size_t)(t * NBATCH + b) * KTOTC;
  const float* wo = ksrow + SEGC + g * CGRP * CGRP;
  const float* g1 = G1 + (size_t)((b * NGRP + g) * CGRP) * HWPIX;
  int row0 = mt * 16, col0 = nt * 16;
  v8f c = ZERO8;
  for (int k0 = 0; k0 < CGRP; k0 += 32) {
    v16bf bf = load_b(g1 + k0 * HWPIX + col0, HWPIX, lr, hl);
    v16bf a  = load_a(wo + row0 * CGRP + k0, CGRP, lr, hl);
    c = wmma_bf(a, bf, c);
  }
  const float* boP = ksrow + SEGC + WSZC;
  int col = col0 + lr;
#pragma unroll
  for (int r = 0; r < 8; ++r) {
    int o = row0 + hl * 8 + r;
    size_t oi = ((size_t)(b * NGRP + g) * CGRP + o) * HWPIX + col;
    float dx = c[r] + boP[g * CGRP + o];
    float gate = 1.f / (1.f + __expf(-(XS[oi] + dx)));
    S[oi] = S[oi] * gate;
  }
}

// ------- res_block pass 1: y0=relu(cw0@sg+cb0), xs2=cws@sg+cbs ----------
__global__ void k_res1(const float* __restrict__ SG, const float* __restrict__ cw0,
                       const float* __restrict__ cb0, const float* __restrict__ cws,
                       const float* __restrict__ cbs, float* __restrict__ Y0,
                       float* __restrict__ XS2) {
  int wid  = (blockIdx.x * 256 + threadIdx.x) >> 5;   // 2048 waves
  int lane = threadIdx.x & 31, lr = lane & 15, hl = lane >> 4;
  int nt = wid & 255, mt = (wid >> 8) & 1, b = wid >> 9;
  const float* Bp = SG + (size_t)b * CIN * HWPIX;
  int row0 = mt * 16, col0 = nt * 16;
  v8f c0 = ZERO8, c1 = ZERO8;
  for (int k0 = 0; k0 < CIN; k0 += 32) {
    v16bf bf = load_b(Bp + (size_t)k0 * HWPIX + col0, HWPIX, lr, hl);
    v16bf a0 = load_a(cw0 + row0 * CIN + k0, CIN, lr, hl);
    v16bf a1 = load_a(cws + row0 * CIN + k0, CIN, lr, hl);
    c0 = wmma_bf(a0, bf, c0);
    c1 = wmma_bf(a1, bf, c1);
  }
  int col = col0 + lr;
#pragma unroll
  for (int r = 0; r < 8; ++r) {
    int o = row0 + hl * 8 + r;
    size_t oi = ((size_t)(b * NF + o)) * HWPIX + col;
    Y0[oi]  = fmaxf(c0[r] + cb0[o], 0.f);
    XS2[oi] = c1[r] + cbs[o];
  }
}

// ------- res_block pass 2 + noise + out update ---------------------------
__global__ void k_res2(const float* __restrict__ Y0, const float* __restrict__ XS2,
                       const float* __restrict__ cw1, const float* __restrict__ cb1,
                       const float* __restrict__ nw, const float* __restrict__ nz,
                       const float* __restrict__ outPrev, float* __restrict__ outNext,
                       int t) {
  int wid  = (blockIdx.x * 256 + threadIdx.x) >> 5;   // 2048 waves
  int lane = threadIdx.x & 31, lr = lane & 15, hl = lane >> 4;
  int nt = wid & 255, mt = (wid >> 8) & 1, b = wid >> 9;
  const float* Bp = Y0 + (size_t)b * NF * HWPIX;
  int row0 = mt * 16, col0 = nt * 16;
  v8f c = ZERO8;
  v16bf bf = load_b(Bp + col0, HWPIX, lr, hl);        // K = 32, single step
  v16bf a  = load_a(cw1 + row0 * NF, NF, lr, hl);
  c = wmma_bf(a, bf, c);
  int col = col0 + lr;
  const float* nzp = nz + ((size_t)t * NBATCH + b) * HWPIX;
#pragma unroll
  for (int r = 0; r < 8; ++r) {
    int o = row0 + hl * 8 + r;
    size_t oi = ((size_t)(b * NF + o)) * HWPIX + col;
    float rr = XS2[oi] + c[r] + cb1[o];
    float sfin = rr + nw[t * NF + o] * nzp[col];
    outNext[oi] = outPrev[oi] + 0.1f * sfin;
  }
}

// ---------------- final projection + clip --------------------------------
__global__ void k_final(const float* __restrict__ xf, const float* __restrict__ outW,
                        const float* __restrict__ outb, float* __restrict__ clipD,
                        float* __restrict__ rawD) {
  int idx = blockIdx.x * 256 + threadIdx.x;   // 4*3*4096
  int pix = idx & 4095;
  int c = (idx >> 12) % 3;
  int b = idx / (3 * HWPIX);
  const float* xb = xf + (size_t)b * NF * HWPIX;
  float acc = outb[c];
  for (int f = 0; f < NF; ++f) acc += outW[c * NF + f] * xb[f * HWPIX + pix];
  rawD[idx] = acc;
  clipD[idx] = fminf(fmaxf(acc, -1.f), 1.f);
}

extern "C" void kernel_launch(void* const* d_in, const int* in_sizes, int n_in,
                              void* d_out, int out_size, void* d_ws, size_t ws_size,
                              hipStream_t stream) {
  const float* lat    = (const float*)d_in[0];
  const float* seed   = (const float*)d_in[1];
  const float* latW0  = (const float*)d_in[2];
  const float* latb0  = (const float*)d_in[3];
  const float* latW1  = (const float*)d_in[4];
  const float* latb1  = (const float*)d_in[5];
  const float* dynW1  = (const float*)d_in[6];
  const float* dynb1  = (const float*)d_in[7];
  const float* dynW2  = (const float*)d_in[8];
  const float* dynb2  = (const float*)d_in[9];
  const float* cw0    = (const float*)d_in[10];
  const float* cb0    = (const float*)d_in[11];
  const float* cw1    = (const float*)d_in[12];
  const float* cb1    = (const float*)d_in[13];
  const float* cws    = (const float*)d_in[14];
  const float* cbs    = (const float*)d_in[15];
  const float* noisew = (const float*)d_in[16];
  const float* noise  = (const float*)d_in[17];
  const float* outW   = (const float*)d_in[18];
  const float* outb   = (const float*)d_in[19];

  float* out   = (float*)d_out;
  float* clipD = out;                      // (4,3,64,64)
  float* embs  = out + 49152;              // (7,4,32,64,64)
  float* rawD  = out + 49152 + 7 * SLAB;   // (4,3,64,64)

  float* ws   = (float*)d_ws;
  float* Lseq = ws;                        // 24*512
  float* Hbuf = ws + 12288;                // 32*1024 (rows 24..31 = 0)
  float* KS   = ws + 45056;                // 24*111744
  float* wbuf = ws + 2726912;              // 4*289 sobel kernels
  float* S    = ws + 2728192;              // (4,384,4096), becomes sg in place
  float* G1   = S + 6291456;
  float* XS   = G1 + 6291456;
  float* Y0   = XS + 6291456;              // (4,32,4096)
  float* XS2  = Y0 + SLAB;

  k_sobelw<<<4, 32, 0, stream>>>(wbuf);
  k_latseq<<<1, 512, 0, stream>>>(lat, latW0, latb0, latW1, latb1, Lseq);
  k_hidden<<<128, 256, 0, stream>>>(Lseq, dynW1, dynb1, Hbuf);
  k_ksgemm<<<1746, 256, 0, stream>>>(Hbuf, dynW2, dynb2, KS);
  k_seed<<<2048, 256, 0, stream>>>(seed, embs);

  for (int t = 0; t < NSTEP; ++t) {
    const float* outPrev = embs + (size_t)t * SLAB;
    float* outNext = embs + (size_t)(t + 1) * SLAB;
    k_sobelnorm<<<1536, 256, 0, stream>>>(outPrev, wbuf, S);
    k_gconv1<<<3072, 256, 0, stream>>>(S, KS, G1, XS, t);
    k_gconv2<<<3072, 256, 0, stream>>>(S, G1, XS, KS, t);
    k_res1<<<256, 256, 0, stream>>>(S, cw0, cb0, cws, cbs, Y0, XS2);
    k_res2<<<256, 256, 0, stream>>>(Y0, XS2, cw1, cb1, noisew, noise,
                                    outPrev, outNext, t);
  }
  k_final<<<192, 256, 0, stream>>>(embs + (size_t)6 * SLAB, outW, outb, clipD, rawD);
}
